// LabelCorrelationGCN_42657615184065
// MI455X (gfx1250) — compile-verified, matrix operand order checked by reading, not run on hardware
//
#include <hip/hip_runtime.h>
#include <cstdint>
#include <cstddef>

// ---------------------------------------------------------------------------
// LabelCorrelationGCN for MI455X (gfx1250): 2-layer GCN.
//   GEMM via v_wmma_f32_16x16x32_bf16 with bf16 hi/lo split (fp32-ish acc).
//   Edge gather/scatter: one wave32 per edge, float4 gather + f32 atomics
//   (feature tables are 51MB -> L2-resident on the 192MB L2).
//   Round 3: nt loop kept rolled (#pragma unroll 1) so only ~110 VGPRs are
//   live -> no scratch spills, no s_set_vgpr_msb; full-tile fast path for
//   A-loads and stores (N % 16 == 0 => guards almost never execute).
// ---------------------------------------------------------------------------

typedef __attribute__((ext_vector_type(16))) __bf16 v16bf;
typedef __attribute__((ext_vector_type(8)))  float  v8f;

#define DIM 128

__device__ __forceinline__ void split_bf16(float x, __bf16& hi, __bf16& lo) {
    hi = (__bf16)x;
    lo = (__bf16)(x - (float)hi);
}

// ---------------- degree / norm prep ----------------

__global__ void k_deg_init(float* __restrict__ deg, int n) {
    int i = blockIdx.x * blockDim.x + threadIdx.x;
    if (i < n) deg[i] = 1.0f;            // self-loop weight
}

__global__ void k_deg_accum(float* __restrict__ deg,
                            const int* __restrict__ ei,
                            const float* __restrict__ ew, int n_edges) {
    int e = blockIdx.x * blockDim.x + threadIdx.x;
    if (e < n_edges) atomicAdd(&deg[ei[n_edges + e]], ew[e]);   // col index
}

__global__ void k_rsqrt_inplace(float* __restrict__ deg, int n) {
    int i = blockIdx.x * blockDim.x + threadIdx.x;
    if (i < n) {
        float d = deg[i];
        deg[i] = (d > 0.0f) ? rsqrtf(fmaxf(d, 1e-38f)) : 0.0f;
    }
}

__global__ void k_edge_norm(float* __restrict__ nrm,
                            const int* __restrict__ ei,
                            const float* __restrict__ ew,
                            const float* __restrict__ dinv,
                            int n_edges, int n_nodes) {
    int e = blockIdx.x * blockDim.x + threadIdx.x;
    int total = n_edges + n_nodes;
    if (e >= total) return;
    if (e < n_edges) {
        int r = ei[e], c = ei[n_edges + e];
        nrm[e] = dinv[r] * ew[e] * dinv[c];
    } else {
        float di = dinv[e - n_edges];
        nrm[e] = di * di;                // self-loop, weight 1
    }
}

// ---------------- bias broadcast init ----------------

__global__ void k_bias_init(float* __restrict__ out, const float* __restrict__ b, int total) {
    int i = blockIdx.x * blockDim.x + threadIdx.x;
    if (i < total) out[i] = b[i & (DIM - 1)];
}

// ---------------- WMMA GEMM: H[n_rows,128] = relu?(X)[n_rows,128] @ W[128,128] ----
// One wave computes a 16-row x 128-col stripe. A fragments (hi/lo, all 4
// k-tiles) are built once per stripe. The nt loop is kept ROLLED with a single
// live v8f accumulator; B fragments are loaded from LDS and consumed
// immediately; each 16x16 tile is stored as soon as it completes.
// 96 WMMAs/stripe. W^T (bf16 hi/lo) staged in LDS per block (2 * 32KB).

__global__ __launch_bounds__(256, 4)
void k_gemm_wmma(const float* __restrict__ X, const float* __restrict__ W,
                 float* __restrict__ H, int n_rows, int do_relu) {
    __shared__ __bf16 WT_hi[DIM * DIM];   // WT[n][k] = W[k][n]
    __shared__ __bf16 WT_lo[DIM * DIM];

    for (int idx = threadIdx.x; idx < DIM * DIM; idx += blockDim.x) {
        int k = idx >> 7, n = idx & (DIM - 1);
        __bf16 hi, lo;
        split_bf16(W[idx], hi, lo);
        WT_hi[n * DIM + k] = hi;
        WT_lo[n * DIM + k] = lo;
    }
    __syncthreads();

    const int lane   = threadIdx.x & 31;
    const int nlane  = lane & 15;         // N index (B/C/D) and M row (A loads)
    const int khalf  = lane >> 4;         // which K-half this lane holds
    const int wgid   = (blockIdx.x * blockDim.x + threadIdx.x) >> 5;
    const int nwaves = (gridDim.x * blockDim.x) >> 5;
    const int n_mtiles = (n_rows + 15) >> 4;

    for (int mt = wgid; mt < n_mtiles; mt += nwaves) {
        const int  m    = mt * 16 + nlane;
        const bool full = (mt * 16 + 16 <= n_rows);   // true for every tile when N%16==0
        const float* xr = X + (size_t)m * DIM;

        // ---- build all A fragments once (4 k-tiles, hi/lo) ----
        v16bf a_hi[4], a_lo[4];
        #pragma unroll
        for (int kt = 0; kt < 4; ++kt) {
            // lane (m=nlane, half=khalf) holds
            // x[m][kt*32 + khalf*8 + 0..7] and x[m][kt*32 + 16 + khalf*8 + 0..7]
            float c[16];
            if (full || m < n_rows) {
                const float* base = xr + kt * 32 + khalf * 8;
                float4 p0 = *(const float4*)(base + 0);
                float4 p1 = *(const float4*)(base + 4);
                float4 q0 = *(const float4*)(base + 16);
                float4 q1 = *(const float4*)(base + 20);
                c[0]=p0.x;  c[1]=p0.y;  c[2]=p0.z;  c[3]=p0.w;
                c[4]=p1.x;  c[5]=p1.y;  c[6]=p1.z;  c[7]=p1.w;
                c[8]=q0.x;  c[9]=q0.y;  c[10]=q0.z; c[11]=q0.w;
                c[12]=q1.x; c[13]=q1.y; c[14]=q1.z; c[15]=q1.w;
            } else {
                #pragma unroll
                for (int j = 0; j < 16; ++j) c[j] = 0.0f;
            }
            if (do_relu) {
                #pragma unroll
                for (int j = 0; j < 16; ++j) c[j] = fmaxf(c[j], 0.0f);
            }
            #pragma unroll
            for (int j = 0; j < 16; ++j) {
                __bf16 h_, l_;
                split_bf16(c[j], h_, l_);
                a_hi[kt][j] = h_;
                a_lo[kt][j] = l_;
            }
        }

        // ---- per output n-tile: accumulate over k, store immediately ----
        // Rolled loop: keeps only one accumulator + one B hi/lo pair live.
        #pragma unroll 1
        for (int nt = 0; nt < 8; ++nt) {
            v8f acc = (v8f){};
            const __bf16* bh_row = WT_hi + (size_t)(nt * 16 + nlane) * DIM + khalf * 8;
            const __bf16* bl_row = WT_lo + (size_t)(nt * 16 + nlane) * DIM + khalf * 8;
            #pragma unroll
            for (int kt = 0; kt < 4; ++kt) {
                const __bf16* bh = bh_row + kt * 32;
                const __bf16* bl = bl_row + kt * 32;
                v16bf b_hi, b_lo;
                #pragma unroll
                for (int j = 0; j < 8; ++j) {
                    b_hi[j]     = bh[j];
                    b_hi[8 + j] = bh[16 + j];
                    b_lo[j]     = bl[j];
                    b_lo[8 + j] = bl[16 + j];
                }
                acc = __builtin_amdgcn_wmma_f32_16x16x32_bf16(
                    false, a_hi[kt], false, b_hi, (short)0, acc, false, false);
                acc = __builtin_amdgcn_wmma_f32_16x16x32_bf16(
                    false, a_hi[kt], false, b_lo, (short)0, acc, false, false);
                acc = __builtin_amdgcn_wmma_f32_16x16x32_bf16(
                    false, a_lo[kt], false, b_hi, (short)0, acc, false, false);
            }
            // D layout: VGPR r, lanes 0-15 -> M = r, lanes 16-31 -> M = r + 8
            float* o = H + (size_t)(mt * 16 + 8 * khalf) * DIM + nt * 16 + nlane;
            if (full) {
                #pragma unroll
                for (int r = 0; r < 8; ++r)
                    o[(size_t)r * DIM] = acc[r];      // constant 512B offsets
            } else {
                #pragma unroll
                for (int r = 0; r < 8; ++r) {
                    if (mt * 16 + r + 8 * khalf < n_rows)
                        o[(size_t)r * DIM] = acc[r];
                }
            }
        }
    }
}

// ---------------- edge gather/scale/scatter-add ----------------
// One wave per edge (incl. self-loops). Lane j owns dims 4j..4j+3:
// 128-bit gather from h[row], scale by norm, 4x global_atomic_add_f32 to out[col].

__global__ __launch_bounds__(256)
void k_scatter(const float* __restrict__ H, const int* __restrict__ ei,
               const float* __restrict__ nrm, float* __restrict__ out,
               int n_edges, int n_nodes) {
    const int lane   = threadIdx.x & 31;
    int w            = (blockIdx.x * blockDim.x + threadIdx.x) >> 5;
    const int nwaves = (gridDim.x * blockDim.x) >> 5;
    const int total  = n_edges + n_nodes;

    for (int e = w; e < total; e += nwaves) {
        int r, c;
        if (e < n_edges) { r = ei[e]; c = ei[n_edges + e]; }
        else             { r = e - n_edges; c = r; }

        // hint the next source row toward L2/L0 (global_prefetch_b8)
        int en = e + nwaves;
        if (en < total) {
            int rn = (en < n_edges) ? ei[en] : (en - n_edges);
            __builtin_prefetch(H + (size_t)rn * DIM + lane * 4, 0, 0);
        }

        float nm = nrm[e];
        float4 hv = *(const float4*)(H + (size_t)r * DIM + lane * 4);
        float* o = out + (size_t)c * DIM + lane * 4;
        atomicAdd(o + 0, hv.x * nm);
        atomicAdd(o + 1, hv.y * nm);
        atomicAdd(o + 2, hv.z * nm);
        atomicAdd(o + 3, hv.w * nm);
    }
}

// ---------------------------------------------------------------------------

static inline size_t align256(size_t x) { return (x + 255) & ~(size_t)255; }

extern "C" void kernel_launch(void* const* d_in, const int* in_sizes, int n_in,
                              void* d_out, int out_size, void* d_ws, size_t ws_size,
                              hipStream_t stream) {
    const float* x   = (const float*)d_in[0];   // [N,128]
    const int*   ei  = (const int*)  d_in[1];   // [2,E] (row-major)
    const float* ew  = (const float*)d_in[2];   // [E]
    const float* W1  = (const float*)d_in[3];   // [128,128]
    const float* b1  = (const float*)d_in[4];   // [128]
    const float* W2  = (const float*)d_in[5];   // [128,128]
    const float* b2  = (const float*)d_in[6];   // [128]
    float* out = (float*)d_out;

    const int N = in_sizes[0] / DIM;
    const int E = in_sizes[1] / 2;
    const int T = E + N;                         // edges incl. self-loops

    char* ws = (char*)d_ws;
    size_t off = 0;
    float* dinv = (float*)(ws + off); off += align256((size_t)N * 4);
    float* nrm  = (float*)(ws + off); off += align256((size_t)T * 4);
    float* h    = (float*)(ws + off); off += align256((size_t)N * DIM * 4);
    float* t1   = (float*)(ws + off); off += align256((size_t)N * DIM * 4);
    (void)ws_size;

    const int BT = 256;
    dim3 blk(BT);

    // ---- degree / normalization ----
    k_deg_init     <<<dim3((N + BT - 1) / BT), blk, 0, stream>>>(dinv, N);
    k_deg_accum    <<<dim3((E + BT - 1) / BT), blk, 0, stream>>>(dinv, ei, ew, E);
    k_rsqrt_inplace<<<dim3((N + BT - 1) / BT), blk, 0, stream>>>(dinv, N);
    k_edge_norm    <<<dim3((T + BT - 1) / BT), blk, 0, stream>>>(nrm, ei, ew, dinv, E, N);

    const int n_mtiles   = (N + 15) / 16;
    const int gemm_blks  = (n_mtiles + 7) / 8;          // 8 waves/block, 1 stripe/wave
    const int scat_blks  = (T + 7) / 8;                  // 8 edges/block (wave32 per edge)
    const int bias_blks  = ((N * DIM) + BT - 1) / BT;

    // ---- layer 1: h = x @ W1 ; t1 = b1 + scatter(norm * h[row]) ----
    k_gemm_wmma<<<dim3(gemm_blks), blk, 0, stream>>>(x, W1, h, N, /*relu=*/0);
    k_bias_init<<<dim3(bias_blks), blk, 0, stream>>>(t1, b1, N * DIM);
    k_scatter  <<<dim3(scat_blks), blk, 0, stream>>>(h, ei, nrm, t1, E, N);

    // ---- layer 2: h = relu(t1) @ W2 ; out = b2 + scatter(norm * h[row]) ----
    k_gemm_wmma<<<dim3(gemm_blks), blk, 0, stream>>>(t1, W2, h, N, /*relu=*/1);
    k_bias_init<<<dim3(bias_blks), blk, 0, stream>>>(out, b2, N * DIM);
    k_scatter  <<<dim3(scat_blks), blk, 0, stream>>>(h, ei, nrm, out, E, N);

    (void)n_in; (void)out_size;
}
